// DGMGMinimal_28887950033186
// MI455X (gfx1250) — compile-verified, matrix operand order checked by reading, not run on hardware
//
#include <hip/hip_runtime.h>

#define HF   64
#define MAXN 64
#define BATCH 64
#define LSEQ 512
#define GPB  16            // graphs per block
#define NBLK (BATCH / GPB) // 4 workgroups
#define NTHR 256           // 8 waves (wave32)

typedef __bf16 bf16x8  __attribute__((ext_vector_type(8)));
typedef __bf16 bf16x16 __attribute__((ext_vector_type(16)));
typedef float  f32x8   __attribute__((ext_vector_type(8)));

__device__ __forceinline__ __bf16 f2bf(float f) {
    unsigned u = __builtin_bit_cast(unsigned, f);
    unsigned r = u + 0x7FFFu + ((u >> 16) & 1u);   // round-to-nearest-even
    unsigned short h = (unsigned short)(r >> 16);
    return __builtin_bit_cast(__bf16, h);
}
__device__ __forceinline__ float sigm(float x)   { return 1.f / (1.f + __expf(-x)); }
__device__ __forceinline__ float tanh_f(float x) { return 2.f / (1.f + __expf(-2.f * x)) - 1.f; }
__device__ __forceinline__ int   iclamp(int v, int lo, int hi) { return v < lo ? lo : (v > hi ? hi : v); }

// A/B fragment loader: 16 bf16 per lane, two contiguous 8-element runs (ds_load_b128 x2)
__device__ __forceinline__ bf16x16 load_frag(const __bf16* rowp, int ka, int kb) {
    bf16x8 lo = *(const bf16x8*)(rowp + ka);
    bf16x8 hi = *(const bf16x8*)(rowp + kb);
    return __builtin_shufflevector(lo, hi, 0,1,2,3,4,5,6,7,8,9,10,11,12,13,14,15);
}

// One 16(M=graphs) x 16(N) output tile of C = A(16xK) @ B(NxK)^T + bias, optional relu.
__device__ __forceinline__ void mma_tile(const __bf16* A, int lda, const __bf16* B,
                                         float* C, int ldc, const float* bias,
                                         int K, int n0, int relu) {
    const int lane = threadIdx.x & 31;
    const int r    = lane & 15;   // M for A, N for B/D-lane
    const int hh   = lane >> 4;   // K-half selector
    f32x8 acc = {0.f, 0.f, 0.f, 0.f, 0.f, 0.f, 0.f, 0.f};
    for (int k0 = 0; k0 < K; k0 += 32) {
        bf16x16 av = load_frag(A + r * lda,        k0 + hh * 8, k0 + 16 + hh * 8);
        bf16x16 bv = load_frag(B + (n0 + r) * K,   k0 + hh * 8, k0 + 16 + hh * 8);
        acc = __builtin_amdgcn_wmma_f32_16x16x32_bf16(false, av, false, bv,
                                                      (short)0, acc, false, false);
    }
    float bi = bias[n0 + r];
#pragma unroll
    for (int rr = 0; rr < 8; ++rr) {
        float v = acc[rr] + bi;
        if (relu) v = fmaxf(v, 0.f);
        C[(rr + hh * 8) * ldc + n0 + r] = v;   // D: N=lane&15, M=vgpr+8*(lane>>4)
    }
}

__device__ __forceinline__ void cp_bf(__bf16* dst, const float* src, int n) {
    for (int i = threadIdx.x; i < n; i += NTHR) dst[i] = f2bf(src[i]);
}
__device__ __forceinline__ void cp_f(float* dst, const float* src, int n) {
    for (int i = threadIdx.x; i < n; i += NTHR) dst[i] = src[i];
}

__global__ __launch_bounds__(NTHR)
void dgmg_kernel(const int* __restrict__ acts, const int* __restrict__ dsts,
                 const int* __restrict__ lens,
                 const float* __restrict__ node_init, const float* __restrict__ Wproj,
                 const float* __restrict__ bproj,
                 const float* __restrict__ nWih, const float* __restrict__ nWhh,
                 const float* __restrict__ nbih, const float* __restrict__ nbhh,
                 const float* __restrict__ gWih, const float* __restrict__ gWhh,
                 const float* __restrict__ gbih, const float* __restrict__ gbhh,
                 const float* __restrict__ nmW1, const float* __restrict__ nmb1,
                 const float* __restrict__ nmW2, const float* __restrict__ nmb2,
                 const float* __restrict__ emW1, const float* __restrict__ emb1,
                 const float* __restrict__ emW2, const float* __restrict__ emb2,
                 const float* __restrict__ Wq,   const float* __restrict__ bq,
                 const float* __restrict__ Wk,   const float* __restrict__ bk,
                 const float* __restrict__ graph0,
                 float* __restrict__ out, float* __restrict__ nodesG) {
    // ---- LDS-resident bf16 weights ----
    __shared__ __bf16 s_nWih[192 * 64], s_nWhh[192 * 64], s_gWih[192 * 64], s_gWhh[192 * 64];
    __shared__ __bf16 s_nmW1[64 * 64], s_emW1[64 * 128], s_Wq[64 * 128], s_WkT[64 * 64];
    // bf16 activations
    __shared__ __bf16 sGCbf[GPB * 128], sCURbf[GPB * 64], sXDbf[GPB * 64],
                      sXINbf[GPB * 64], sQbf[GPB * 64];
    // f32 activations / gates
    __shared__ float sGI[GPB * 192], sGH[GPB * 192];
    __shared__ float sG[GPB * 64], sCURf[GPB * 64], sT1[GPB * 64], sT2[GPB * 64],
                     sQ[GPB * 64], sWQ[GPB * 64], sXIN[GPB * 64], sGNEW[GPB * 64],
                     sDL[GPB * 64];
    // f32 small weights / biases
    __shared__ float s_nmW2[2 * 64], s_emW2[2 * 64], s_nbih[192], s_nbhh[192],
                     s_gbih[192], s_gbhh[192], s_nmb1[64], s_emb1[64], s_bq[64],
                     s_bk[64], s_nmb2[2], s_emb2[2], sH0[64], s_zb[64];
    // per-graph state
    __shared__ float sNll[GPB], sNL[GPB * 2], sEL[GPB * 2], sBKQ[GPB];
    __shared__ int sCur[GPB], sDone[GPB], sAct[GPB], sDsafe[GPB], sActive[GPB],
                   sUpd[GPB], sIdx[GPB], sLenv[GPB];

    const int tid  = threadIdx.x;
    const int g0   = blockIdx.x * GPB;
    // Scalar wave id: tile dispatch becomes SALU compares + scalar branches,
    // never touching EXEC (WMMA requires EXEC == all ones).
    const int wave = __builtin_amdgcn_readfirstlane(tid >> 5);

    // ---------------- init ----------------
    {
        float* nb = nodesG + (size_t)g0 * MAXN * HF;
        for (int i = tid; i < GPB * MAXN * HF; i += NTHR) nb[i] = 0.f;
    }
    cp_bf(s_nWih, nWih, 192 * 64);  cp_bf(s_nWhh, nWhh, 192 * 64);
    cp_bf(s_gWih, gWih, 192 * 64);  cp_bf(s_gWhh, gWhh, 192 * 64);
    cp_bf(s_nmW1, nmW1, 64 * 64);   cp_bf(s_emW1, emW1, 64 * 128);
    cp_bf(s_Wq,   Wq,   64 * 128);
    for (int i = tid; i < 64 * 64; i += NTHR) { // transpose Wk -> WkT[c*64+j] = Wk[j][c]
        int j = i >> 6, k = i & 63;
        s_WkT[k * 64 + j] = f2bf(Wk[i]);
    }
    cp_f(s_nmW2, nmW2, 2 * 64); cp_f(s_emW2, emW2, 2 * 64);
    cp_f(s_nbih, nbih, 192);    cp_f(s_nbhh, nbhh, 192);
    cp_f(s_gbih, gbih, 192);    cp_f(s_gbhh, gbhh, 192);
    cp_f(s_nmb1, nmb1, 64);     cp_f(s_emb1, emb1, 64);
    cp_f(s_bq, bq, 64);         cp_f(s_bk, bk, 64);
    cp_f(s_nmb2, nmb2, 2);      cp_f(s_emb2, emb2, 2);
    if (tid < 64) {
        s_zb[tid] = 0.f;
        // h0 = Wproj @ node_init + bproj
        float acc = bproj[tid];
        for (int d = 0; d < 64; ++d) acc += node_init[d] * Wproj[tid * 64 + d];
        sH0[tid] = acc;
    }
    if (tid < GPB) {
        sCur[tid] = -1; sDone[tid] = 0; sNll[tid] = 0.f;
        sLenv[tid] = lens[g0 + tid];
    }
    for (int e = tid; e < GPB * HF; e += NTHR) sG[e] = graph0[e & 63];
    __syncthreads();

    // ---------------- sequential steps ----------------
    for (int t = 0; t < LSEQ; ++t) {
        // P0: flags + gathers + bf16 staging
        if (tid < GPB) {
            int a = acts[(g0 + tid) * LSEQ + t];
            int d = dsts[(g0 + tid) * LSEQ + t];
            sAct[tid]    = a;
            sDsafe[tid]  = iclamp(d, 0, MAXN - 1);
            sActive[tid] = (!sDone[tid]) && (a >= 0) && (t < sLenv[tid]);
        }
        for (int e = tid; e < GPB * HF; e += NTHR) {
            int b = e >> 6, k = e & 63;
            int cs  = iclamp(sCur[b], 0, MAXN - 1);
            int ds2 = iclamp(dsts[(g0 + b) * LSEQ + t], 0, MAXN - 1);
            const float* nb = nodesG + (size_t)(g0 + b) * MAXN * HF;
            float cv = nb[cs * HF + k];
            float xv = nb[ds2 * HF + k];
            sCURf[e]  = cv;
            sCURbf[e] = f2bf(cv);
            sXDbf[e]  = f2bf(xv);
            float gv  = sG[e];
            sGCbf[b * 128 + k]      = f2bf(gv);
            sGCbf[b * 128 + 64 + k] = f2bf(cv);
        }
        __syncthreads();

        // P1: 36 WMMA tiles: T1, T2, Q, node-GRU gi/gh
        for (int ti = wave; ti < 36; ti += 8) {
            if (ti < 4)       mma_tile(sGCbf, 128, s_nmW1, sT1, 64,  s_nmb1, 64,  ti * 16,        1);
            else if (ti < 8)  mma_tile(sGCbf, 128, s_emW1, sT2, 64,  s_emb1, 128, (ti - 4) * 16,  1);
            else if (ti < 12) mma_tile(sGCbf, 128, s_Wq,   sQ,  64,  s_bq,   128, (ti - 8) * 16,  0);
            else if (ti < 24) mma_tile(sXDbf, 64,  s_nWih, sGI, 192, s_nbih, 64,  (ti - 12) * 16, 0);
            else              mma_tile(sCURbf,64,  s_nWhh, sGH, 192, s_nbhh, 64,  (ti - 24) * 16, 0);
        }
        __syncthreads();

        // P2a: node-GRU elementwise -> XIN; nl/el heads; bk.q; bf16 staging
        for (int e = tid; e < GPB * HF; e += NTHR) {
            int b = e >> 6, k = e & 63;
            float r = sigm(sGI[b * 192 + k] + sGH[b * 192 + k]);
            float z = sigm(sGI[b * 192 + 64 + k] + sGH[b * 192 + 64 + k]);
            float n = tanh_f(sGI[b * 192 + 128 + k] + r * sGH[b * 192 + 128 + k]);
            float h = sCURf[e];
            float nc = (1.f - z) * n + z * h;
            float xin = (sAct[b] == 0) ? sH0[k] : nc;
            sXIN[e]   = xin;
            sXINbf[e] = f2bf(xin);
            sQbf[e]   = f2bf(sQ[e]);
        }
        if (tid < 64) {
            int b = tid >> 2, w = tid & 3;
            if (w < 2) {
                float acc = s_nmb2[w];
                for (int k = 0; k < HF; ++k) acc += sT1[b * 64 + k] * s_nmW2[w * 64 + k];
                sNL[b * 2 + w] = acc;
            } else {
                int w2 = w - 2;
                float acc = s_emb2[w2];
                for (int k = 0; k < HF; ++k) acc += sT2[b * 64 + k] * s_emW2[w2 * 64 + k];
                sEL[b * 2 + w2] = acc;
            }
        } else if (tid < 80) {
            int b = tid - 64;
            float acc = 0.f;
            for (int k = 0; k < HF; ++k) acc += s_bk[k] * sQ[b * 64 + k];
            sBKQ[b] = acc;
        }
        __syncthreads();

        // P2b: 28 WMMA tiles: graph-GRU gi/gh, wq = Q @ Wk
        for (int ti = wave; ti < 28; ti += 8) {
            if (ti < 12)      mma_tile(sXINbf, 64, s_gWih, sGI, 192, s_gbih, 64, ti * 16,        0);
            else if (ti < 24) mma_tile(sGCbf, 128, s_gWhh, sGH, 192, s_gbhh, 64, (ti - 12) * 16, 0);
            else              mma_tile(sQbf,   64, s_WkT,  sWQ, 64,  s_zb,   64, (ti - 24) * 16, 0);
        }
        __syncthreads();

        // P3: graph-GRU elementwise -> GNEW ; dl[i] = nodes[i].wq + bk.q
        for (int e = tid; e < GPB * HF; e += NTHR) {
            int b = e >> 6, k = e & 63;
            float r = sigm(sGI[b * 192 + k] + sGH[b * 192 + k]);
            float z = sigm(sGI[b * 192 + 64 + k] + sGH[b * 192 + 64 + k]);
            float n = tanh_f(sGI[b * 192 + 128 + k] + r * sGH[b * 192 + 128 + k]);
            float h = sG[e];
            sGNEW[e] = (1.f - z) * n + z * h;
        }
        for (int dot = tid; dot < GPB * MAXN; dot += NTHR) {
            int b = dot >> 6, i = dot & 63;
            const float4* np = (const float4*)(nodesG + ((size_t)(g0 + b) * MAXN + i) * HF);
            const float* w = sWQ + b * 64;
            float s = sBKQ[b];
#pragma unroll
            for (int k4 = 0; k4 < 16; ++k4) {
                float4 v = np[k4];
                s += v.x * w[k4 * 4 + 0] + v.y * w[k4 * 4 + 1] +
                     v.z * w[k4 * 4 + 2] + v.w * w[k4 * 4 + 3];
            }
            sDL[dot] = (i < sCur[b]) ? s : -1.0e9f;
        }
        __syncthreads();

        // P4: per-graph logsumexp, nll, state transition
        if (tid < GPB) {
            int b = tid;
            float l0 = sNL[b * 2], l1 = sNL[b * 2 + 1];
            float mn = fmaxf(l0, l1);
            float lse_n = mn + __logf(__expf(l0 - mn) + __expf(l1 - mn));
            float e0 = sEL[b * 2], e1 = sEL[b * 2 + 1];
            float me = fmaxf(e0, e1);
            float lse_e = me + __logf(__expf(e0 - me) + __expf(e1 - me));
            float m = -3.4e38f;
            for (int i = 0; i < MAXN; ++i) m = fmaxf(m, sDL[b * 64 + i]);
            float ssum = 0.f;
            for (int i = 0; i < MAXN; ++i) ssum += __expf(sDL[b * 64 + i] - m);
            float ce = (m + __logf(ssum)) - sDL[b * 64 + sDsafe[b]];
            int a = sAct[b];
            float st = (a == 0) ? lse_n - l0 : (a == 1) ? lse_n - l1 :
                       (a == 2) ? lse_e - e0 : (a == 3) ? lse_e - e1 : ce;
            if (sActive[b]) sNll[b] += st;
            int is_addn = (a == 0), is_dest = (a == 4), is_stopn = (a == 1);
            int cur = sCur[b];
            sUpd[b] = sActive[b] && (is_addn || is_dest);
            sIdx[b] = is_addn ? iclamp(cur + 1, 0, MAXN - 1) : iclamp(cur, 0, MAXN - 1);
            if (sActive[b] && is_addn) sCur[b] = cur + 1;
            if (a < 0 || (sActive[b] && is_stopn)) sDone[b] = 1;
        }
        __syncthreads();

        // P5: commit nodes row + graph state where upd
        for (int e = tid; e < GPB * HF; e += NTHR) {
            int b = e >> 6, k = e & 63;
            if (sUpd[b]) {
                nodesG[((size_t)(g0 + b) * MAXN + sIdx[b]) * HF + k] = sXIN[e];
                sG[e] = sGNEW[e];
            }
        }
        __syncthreads();
    }

    if (tid == 0) {
        float s = 0.f;
        for (int b = 0; b < GPB; ++b) s += sNll[b];
        atomicAdd(out, s);
    }
}

__global__ void zero_out_kernel(float* o) { o[0] = 0.f; }

extern "C" void kernel_launch(void* const* d_in, const int* in_sizes, int n_in,
                              void* d_out, int out_size, void* d_ws, size_t ws_size,
                              hipStream_t stream) {
    (void)in_sizes; (void)n_in; (void)out_size; (void)ws_size;
    const int*   acts      = (const int*)d_in[0];
    const int*   dsts      = (const int*)d_in[1];
    const int*   lens      = (const int*)d_in[2];
    const float* node_init = (const float*)d_in[3];
    const float* Wproj     = (const float*)d_in[4];
    const float* bproj     = (const float*)d_in[5];
    const float* nWih      = (const float*)d_in[6];
    const float* nWhh      = (const float*)d_in[7];
    const float* nbih      = (const float*)d_in[8];
    const float* nbhh      = (const float*)d_in[9];
    const float* gWih      = (const float*)d_in[10];
    const float* gWhh      = (const float*)d_in[11];
    const float* gbih      = (const float*)d_in[12];
    const float* gbhh      = (const float*)d_in[13];
    const float* nmW1      = (const float*)d_in[14];
    const float* nmb1      = (const float*)d_in[15];
    const float* nmW2      = (const float*)d_in[16];
    const float* nmb2      = (const float*)d_in[17];
    const float* emW1      = (const float*)d_in[18];
    const float* emb1      = (const float*)d_in[19];
    const float* emW2      = (const float*)d_in[20];
    const float* emb2      = (const float*)d_in[21];
    const float* Wq        = (const float*)d_in[22];
    const float* bq        = (const float*)d_in[23];
    const float* Wk        = (const float*)d_in[24];
    const float* bk        = (const float*)d_in[25];
    const float* graph0    = (const float*)d_in[26];
    float* out    = (float*)d_out;
    float* nodesW = (float*)d_ws;   // 64*64*64 f32 = 1 MiB scratch

    zero_out_kernel<<<1, 1, 0, stream>>>(out);
    dgmg_kernel<<<NBLK, NTHR, 0, stream>>>(
        acts, dsts, lens, node_init, Wproj, bproj,
        nWih, nWhh, nbih, nbhh, gWih, gWhh, gbih, gbhh,
        nmW1, nmb1, nmW2, nmb2, emW1, emb1, emW2, emb2,
        Wq, bq, Wk, bk, graph0, out, nodesW);
}
